// MSTF_1082331758652
// MI455X (gfx1250) — matrix operand order repaired and checked by reading, not compiled, not run on hardware
//
#include <hip/hip_runtime.h>
#include <hip/hip_bf16.h>

// CDNA5 / gfx1250 correlation-pyramid lookup (RAFT-style).
// Per (batch, pixel) at a compile-time pyramid level: 100 grid dot-products
// via v_wmma_f32_16x16x32_f16 (M = grid points, K = channels, N padded by
// broadcasting f1 into all 16 B columns), then masked bilinear combination of
// the 81 outputs. Level is a template parameter so gather strides fold into
// global_load immediate offsets; loop-carried state is kept as integer
// offsets off the kernarg pointers so address-space inference keeps the
// gathers on the global_load path (not flat_load, which also ticks DScnt).

typedef __attribute__((ext_vector_type(16))) _Float16 v16h;
typedef __attribute__((ext_vector_type(8)))  float    v8f;

#define RADIUS  4
#define KOUT    81     // (2*RADIUS+1)^2
#define GRID    10     // integer gather-grid side (2*RADIUS+2)
#define NPTS    100    // GRID*GRID
#define MTILES  7      // ceil(NPTS/16)
#define CCH     256
#define HF      64
#define WF      64
#define LVLS    4

template <int LVL>
__global__ __launch_bounds__(256)
void MSTF_corr_wmma_kernel(const float* __restrict__ fmap1,
                           const float* __restrict__ f2,
                           const float* __restrict__ coords,
                           float* __restrict__ out)
{
    constexpr int H2  = HF >> LVL;
    constexpr int W2  = WF >> LVL;
    constexpr int CS2 = H2 * W2;      // f2 channel stride (compile-time)
    constexpr int CS1 = HF * WF;      // f1 channel stride

    __shared__ float Dlds[8][112];    // per-wave D values (100 used)

    const int lane = threadIdx.x & 31;
    const int wid  = threadIdx.x >> 5;
    const int task = blockIdx.x * 8 + wid;   // [0, 2*4096)
    const int pix  = task & 4095;
    const int h    = pix >> 6;
    const int w    = pix & 63;
    const int b    = task >> 12;

    // per-pixel coords at this level's resolution
    const int   pbase = h * WF + w;
    const float inv   = 1.0f / (float)(1 << LVL);
    const float cx = coords[(b * 2 + 0) * CS1 + pbase] * inv;
    const float cy = coords[(b * 2 + 1) * CS1 + pbase] * inv;
    const float fx0 = floorf(cx), fy0 = floorf(cy);
    const int   X0 = (int)fx0,    Y0 = (int)fy0;
    const float fx = cx - fx0,    fy = cy - fy0;

    // A-fragment addressing (16-bit 16x32 A layout):
    //  lane L holds row M = L&15; halves 0..7 -> K {0..7}, 8..15 -> K {16..23};
    //  lanes >= 16 add +8 to K (folded into the base offset below).
    const int m   = lane & 15;
    const int hiK = (lane >= 16) ? 8 : 0;

    // Per-tile A row base offsets (elements). Grid point clamped to the
    // feature map; rows 100..111 are padding and masked out in the combine.
    int aOff[MTILES];
    #pragma unroll
    for (int t = 0; t < MTILES; ++t) {
        int g = t * 16 + m;
        if (g > NPTS - 1) g = NPTS - 1;
        const int gy = g / GRID, gx = g - gy * GRID;
        int yc = Y0 - RADIUS + gy; yc = yc < 0 ? 0 : (yc > H2 - 1 ? H2 - 1 : yc);
        int xc = X0 - RADIUS + gx; xc = xc < 0 ? 0 : (xc > W2 - 1 ? W2 - 1 : xc);
        aOff[t] = (b * CCH + hiK) * CS2 + yc * W2 + xc;
    }
    // B base offset (same K order as A: halves 0..7 -> K {0..7}, 8..15 ->
    // K {16..23}; lanes >= 16 add +8, folded into base).
    int bOff = (b * CCH + ((lane >= 16) ? 8 : 0)) * CS1 + pbase;

    v8f acc[MTILES];
    #pragma unroll
    for (int t = 0; t < MTILES; ++t) acc[t] = (v8f){0, 0, 0, 0, 0, 0, 0, 0};

    for (int kb = 0; kb < CCH / 32; ++kb) {
        // B fragment: f1 vector broadcast into all 16 columns.
        v16h Bv;
        #pragma unroll
        for (int hh = 0; hh < 16; ++hh)
            Bv[hh] = (_Float16)fmap1[bOff + ((hh < 8) ? hh : hh + 8) * CS1];   // const imm offsets
        bOff += 32 * CS1;

        #pragma unroll
        for (int t = 0; t < MTILES; ++t) {
            v16h Av;
            #pragma unroll
            for (int hh = 0; hh < 16; ++hh) {
                const int K = (hh < 8) ? hh : hh + 8;    // compile-time
                Av[hh] = (_Float16)f2[aOff[t] + K * CS2]; // const imm offsets
            }
            aOff[t] += 32 * CS2;

            acc[t] = __builtin_amdgcn_wmma_f32_16x16x32_f16(
                false, Av, false, Bv, (short)0, acc[t], false, false);
        }
    }

    // D layout: acc[t][v] at lane l is D[t*16 + v + (l>=16 ? 8 : 0)] for column
    // N = l&15 (all columns identical). Lanes 0 and 16 cover all 16 rows/tile.
    if (lane == 0 || lane == 16) {
        const int mo = (lane == 16) ? 8 : 0;
        #pragma unroll
        for (int t = 0; t < MTILES; ++t)
            #pragma unroll
            for (int v = 0; v < 8; ++v)
                Dlds[wid][t * 16 + mo + v] = acc[t][v];
    }
    __syncthreads();

    // Masked bilinear combination: corr[dy,dx] = sum over 4 corners of the
    // 10x10 grid, weight zeroed outside [0,W2-1]x[0,H2-1], scaled by 1/sqrt(C).
    const float scale = 0.0625f;                 // 1/sqrt(256)
    const float wx[2] = {1.0f - fx, fx};
    const float wy[2] = {1.0f - fy, fy};
    for (int o = lane; o < KOUT; o += 32) {
        const int dyI = o / 9, dxI = o - dyI * 9;
        float s = 0.0f;
        #pragma unroll
        for (int cyi = 0; cyi < 2; ++cyi) {
            const int py = Y0 - RADIUS + dyI + cyi;
            const bool vy = (py >= 0) && (py <= H2 - 1);
            #pragma unroll
            for (int cxi = 0; cxi < 2; ++cxi) {
                const int px = X0 - RADIUS + dxI + cxi;
                const bool vx = (px >= 0) && (px <= W2 - 1);
                const float wgt = (vx && vy) ? wx[cxi] * wy[cyi] : 0.0f;
                s += wgt * Dlds[wid][(dyI + cyi) * GRID + (dxI + cxi)];
            }
        }
        out[((b * (LVLS * KOUT) + LVL * KOUT + o) * HF + h) * WF + w] = s * scale;
    }
}

extern "C" void kernel_launch(void* const* d_in, const int* in_sizes, int n_in,
                              void* d_out, int out_size, void* d_ws, size_t ws_size,
                              hipStream_t stream) {
    const float* fmap1  = (const float*)d_in[0];
    const float* f2_0   = (const float*)d_in[1];
    const float* f2_1   = (const float*)d_in[2];
    const float* f2_2   = (const float*)d_in[3];
    const float* f2_3   = (const float*)d_in[4];
    const float* coords = (const float*)d_in[5];
    float* out = (float*)d_out;

    // 2 batches * 4096 pixels = 8192 wave-tasks per level, 8 waves per block.
    MSTF_corr_wmma_kernel<0><<<1024, 256, 0, stream>>>(fmap1, f2_0, coords, out);
    MSTF_corr_wmma_kernel<1><<<1024, 256, 0, stream>>>(fmap1, f2_1, coords, out);
    MSTF_corr_wmma_kernel<2><<<1024, 256, 0, stream>>>(fmap1, f2_2, coords, out);
    MSTF_corr_wmma_kernel<3><<<1024, 256, 0, stream>>>(fmap1, f2_3, coords, out);
}
